// GCN_43611097924207
// MI455X (gfx1250) — compile-verified, matrix operand order checked by reading, not run on hardware
//
#include <hip/hip_runtime.h>

// ---------------------------------------------------------------------------
// 2-layer GCN for MI455X (gfx1250, wave32).
//   deg/dinv  -> WMMA GEMM (f16 in, f32 acc) -> init(self-loop+bias)
//   -> edge scatter (f32 atomics) -> relu -> repeat for layer 2.
// Scatter is the HBM-bound phase (~1.8 GB total @ 23.3 TB/s); GEMMs ride the
// v_wmma_f32_16x16x32_f16 pipe with weights staged in LDS.
// ---------------------------------------------------------------------------

typedef __attribute__((ext_vector_type(16))) _Float16 v16h;
typedef __attribute__((ext_vector_type(8)))  float    v8f;

// ------------------------------- utility -----------------------------------

__global__ void zero_f32_kernel(float* __restrict__ p, int n) {
    for (int i = blockIdx.x * blockDim.x + threadIdx.x; i < n;
         i += gridDim.x * blockDim.x)
        p[i] = 0.0f;
}

__global__ void degree_kernel(const int* __restrict__ dst, int E,
                              float* __restrict__ deg, int n) {
    for (int e = blockIdx.x * blockDim.x + threadIdx.x; e < E;
         e += gridDim.x * blockDim.x) {
        int d = dst[e];
        if ((unsigned)d < (unsigned)n) atomicAdd(&deg[d], 1.0f);
    }
}

// dinv[i] = rsqrt(deg[i] + 1)   (+1 = self loop; always > 0)
__global__ void dinv_kernel(float* __restrict__ deg_dinv, int n) {
    for (int i = blockIdx.x * blockDim.x + threadIdx.x; i < n;
         i += gridDim.x * blockDim.x)
        deg_dinv[i] = rsqrtf(deg_dinv[i] + 1.0f);
}

// ------------------------------ WMMA GEMM ----------------------------------
// H[M x Nc] = X[M x 128] * W[128 x Nc]   (row major, K = 128 fixed)
// Block: (Nc/16) waves; wave w owns column tile w. 8 row tiles per block.
// B fragments (weights) built once per block from LDS; A streamed from global
// with on-the-fly f32 -> f16 conversion.

#define GEMM_K 128
#define ROW_TILES_PER_BLOCK 8

__device__ __forceinline__ int frag_k(int kt, int laneHi, int j) {
    // ISA 7.12.2 16-bit A/B layout: halves 0..7 -> K = base+0..7,
    // halves 8..15 -> K = base+16..23; lane[4] adds +8 to base.
    return kt * 32 + laneHi * 8 + (j & 7) + ((j >> 3) << 4);
}

__global__ void gemm_wmma_kernel(const float* __restrict__ X,
                                 const float* __restrict__ W,
                                 float* __restrict__ H,
                                 int M, int Nc) {
    __shared__ _Float16 Wlds[GEMM_K * 128];  // max 128x128 f16 = 32 KB

    const int tid  = threadIdx.x;
    const int nthr = blockDim.x;

    // Stage W (k-major, row stride Nc) into LDS as f16.
    for (int i = tid; i < GEMM_K * Nc; i += nthr)
        Wlds[i] = (_Float16)W[i];
    __syncthreads();

    const int wave   = tid >> 5;       // column tile index
    const int lane   = tid & 31;
    const int laneLo = lane & 15;
    const int laneHi = lane >> 4;
    const int c0     = wave * 16;

    // Pre-build the 4 K-tile B fragments for this wave's column tile.
    v16h bfrag[4];
#pragma unroll
    for (int kt = 0; kt < 4; ++kt) {
#pragma unroll
        for (int j = 0; j < 16; ++j) {
            int k = frag_k(kt, laneHi, j);
            bfrag[kt][j] = Wlds[k * Nc + c0 + laneLo];
        }
    }

    const int nRowTiles = (M + 15) >> 4;
    for (int rt = 0; rt < ROW_TILES_PER_BLOCK; ++rt) {
        int rowTile = blockIdx.x * ROW_TILES_PER_BLOCK + rt;
        if (rowTile >= nRowTiles) break;              // wave-uniform
        int m0  = rowTile << 4;
        int row = m0 + laneLo;                        // M divisible by 16
        const float* xr = X + (long)row * GEMM_K;

        v8f acc = {0.f, 0.f, 0.f, 0.f, 0.f, 0.f, 0.f, 0.f};
#pragma unroll
        for (int kt = 0; kt < 4; ++kt) {
            v16h afrag;
#pragma unroll
            for (int j = 0; j < 16; ++j)
                afrag[j] = (_Float16)xr[frag_k(kt, laneHi, j)];
            acc = __builtin_amdgcn_wmma_f32_16x16x32_f16(
                false, afrag, false, bfrag[kt], (short)0, acc, false, false);
        }

        // D layout: VGPR r -> rows (m0 + r) [lanes 0-15] / (m0 + r + 8).
        int col = c0 + laneLo;
#pragma unroll
        for (int r = 0; r < 8; ++r)
            H[(long)(m0 + r + laneHi * 8) * Nc + col] = acc[r];
    }
}

// --------------------------- init / scatter / relu -------------------------
// out[i,f] = h[i,f] * dinv[i]^2 + b[f]   (self-loop contribution + bias)
template <int F>
__global__ void init_out_kernel(const float* __restrict__ h,
                                const float* __restrict__ dinv,
                                const float* __restrict__ b,
                                float* __restrict__ out, int n) {
    long total = (long)n * F;
    for (long idx = blockIdx.x * (long)blockDim.x + threadIdx.x; idx < total;
         idx += (long)gridDim.x * blockDim.x) {
        int i = (int)(idx / F);
        int f = (int)(idx - (long)i * F);
        float di = dinv[i];
        out[idx] = h[idx] * di * di + b[f];
    }
}

// One wave per edge; lane owns a contiguous F/32-float chunk.
template <int F>
__global__ void scatter_kernel(const int* __restrict__ src,
                               const int* __restrict__ dst, int E,
                               const float* __restrict__ h,
                               const float* __restrict__ dinv,
                               float* __restrict__ out, int n) {
    constexpr int PER = F / 32;
    const int lane   = threadIdx.x & 31;
    const int gwave  = (blockIdx.x * blockDim.x + threadIdx.x) >> 5;
    const int nwaves = (gridDim.x * blockDim.x) >> 5;

    for (int e = gwave; e < E; e += nwaves) {
        int s = src[e];
        int d = dst[e];
        if ((unsigned)s >= (unsigned)n || (unsigned)d >= (unsigned)n) continue;
        float w = dinv[s] * dinv[d];
        const float* hp = h   + (long)s * F + lane * PER;
        float*       op = out + (long)d * F + lane * PER;
        float v[PER];
#pragma unroll
        for (int j = 0; j < PER; ++j) v[j] = hp[j] * w;   // coalesced b128 load
#pragma unroll
        for (int j = 0; j < PER; ++j) atomicAdd(&op[j], v[j]);
    }
}

__global__ void relu_kernel(float* __restrict__ p, long n) {
    for (long i = blockIdx.x * (long)blockDim.x + threadIdx.x; i < n;
         i += (long)gridDim.x * blockDim.x)
        p[i] = fmaxf(p[i], 0.0f);
}

// --------------------------------- driver ----------------------------------

extern "C" void kernel_launch(void* const* d_in, const int* in_sizes, int n_in,
                              void* d_out, int out_size, void* d_ws, size_t ws_size,
                              hipStream_t stream) {
    const float* x  = (const float*)d_in[0];
    const int*   ei = (const int*)d_in[1];   // [2, E] row-major: src row then dst row
    const float* W1 = (const float*)d_in[2];
    const float* b1 = (const float*)d_in[3];
    const float* W2 = (const float*)d_in[4];
    const float* b2 = (const float*)d_in[5];
    float* out = (float*)d_out;

    const int N = in_sizes[0] / 128;   // 50000 nodes
    const int E = in_sizes[1] / 2;     // 800000 edges
    const int* src = ei;
    const int* dst = ei + E;

    // Workspace layout (floats): dinv[N] | bufA[N*128] | bufB[N*128]
    float* dinv = (float*)d_ws;
    float* bufA = dinv + N;                 // h1, later h2 (N x 64 prefix)
    float* bufB = bufA + (size_t)N * 128;   // a1 = relu(agg(h1))

    const int T = 256;
    const int gN    = (N + T - 1) / T;
    const int gE    = (E + T - 1) / T;
    const int gF128 = 2048;                 // grid-stride over N*128 elements
    const int nRowTiles = (N + 15) / 16;
    const int gGemm = (nRowTiles + ROW_TILES_PER_BLOCK - 1) / ROW_TILES_PER_BLOCK;
    const int gScat = 2048;                 // 16384 waves grid-striding edges

    // ---- normalization: dinv = rsqrt(deg + 1) ----
    zero_f32_kernel<<<gN, T, 0, stream>>>(dinv, N);
    degree_kernel<<<gE, T, 0, stream>>>(dst, E, dinv, N);
    dinv_kernel<<<gN, T, 0, stream>>>(dinv, N);

    // ---- layer 1: h1 = x @ W1 ; a1 = agg(h1) + b1 ; relu ----
    gemm_wmma_kernel<<<gGemm, 256, 0, stream>>>(x, W1, bufA, N, 128);
    init_out_kernel<128><<<gF128, T, 0, stream>>>(bufA, dinv, b1, bufB, N);
    scatter_kernel<128><<<gScat, T, 0, stream>>>(src, dst, E, bufA, dinv, bufB, N);
    relu_kernel<<<gF128, T, 0, stream>>>(bufB, (long)N * 128);

    // ---- layer 2: h2 = a1 @ W2 ; out = agg(h2) + b2 ----
    gemm_wmma_kernel<<<gGemm, 128, 0, stream>>>(bufB, W2, bufA, N, 64);
    init_out_kernel<64><<<gF128, T, 0, stream>>>(bufA, dinv, b2, out, N);
    scatter_kernel<64><<<gScat, T, 0, stream>>>(src, dst, E, bufA, dinv, out, N);
}
